// ClusterMemory_62852551410005
// MI455X (gfx1250) — matrix operand order repaired and checked by reading, not compiled, not run on hardware
//
#include <hip/hip_runtime.h>
#include <hip/hip_bf16.h>
#include <math.h>
#include <stdint.h>

typedef _Float16 half8 __attribute__((ext_vector_type(8)));
typedef _Float16 v16h  __attribute__((ext_vector_type(16)));
typedef float    v8f   __attribute__((ext_vector_type(8)));
typedef uint32_t u32x4 __attribute__((ext_vector_type(4)));
typedef uint32_t u32x8 __attribute__((ext_vector_type(8)));

#define D        2048
#define BATCH    256
#define NFEAT    32768
#define KCLUST   8192
#define INV_TEMP 20.0f
#define BN       64     // N tile per block
#define LDSP     40     // padded LDS row pitch in halves (32 data + 8 pad)

__device__ __forceinline__ v16h combine16(half8 lo, half8 hi) {
  v16h r;
#pragma unroll
  for (int i = 0; i < 8; ++i) { r[i] = lo[i]; r[i + 8] = hi[i]; }
  return r;
}

// ---------------------------------------------------------------------------
// Kernel 0: x = inputs / ||inputs||, stored as f16; zero per-row expsum accum.
// One block per batch row, 256 threads, 8 floats/thread.
// ---------------------------------------------------------------------------
__global__ void normalize_kernel(const float* __restrict__ inputs,
                                 _Float16* __restrict__ xh,
                                 float* __restrict__ sumexp) {
  const int b = blockIdx.x;
  const int t = threadIdx.x;
  const float4* in4 = (const float4*)(inputs + (size_t)b * D);
  float4 a = in4[t * 2 + 0];
  float4 c = in4[t * 2 + 1];
  float ss = a.x*a.x + a.y*a.y + a.z*a.z + a.w*a.w
           + c.x*c.x + c.y*c.y + c.z*c.z + c.w*c.w;
#pragma unroll
  for (int m = 16; m >= 1; m >>= 1) ss += __shfl_xor(ss, m, 32);
  __shared__ float wsum[8];
  __shared__ float rn_s;
  const int lane = t & 31, wv = t >> 5;
  if (lane == 0) wsum[wv] = ss;
  __syncthreads();
  if (t == 0) {
    float s = 0.f;
#pragma unroll
    for (int i = 0; i < 8; ++i) s += wsum[i];
    rn_s = 1.0f / sqrtf(s);
    sumexp[b] = 0.f;
  }
  __syncthreads();
  const float rn = rn_s;
  half8 h;
  h[0]=(_Float16)(a.x*rn); h[1]=(_Float16)(a.y*rn);
  h[2]=(_Float16)(a.z*rn); h[3]=(_Float16)(a.w*rn);
  h[4]=(_Float16)(c.x*rn); h[5]=(_Float16)(c.y*rn);
  h[6]=(_Float16)(c.z*rn); h[7]=(_Float16)(c.w*rn);
  ((half8*)(xh + (size_t)b * D))[t] = h;
}

// ---------------------------------------------------------------------------
// WMMA GEMM:  C[256 x Ntile] = x_f16 @ Bsrc_f32^T  (Bsrc rows are columns)
// Block: 256 threads (8 waves). Block tile M=256, N=64, K-step 32.
// Wave w owns rows [32w, 32w+32): 2 M-subtiles x 4 N-subtiles of 16x16.
// B tile staged through LDS with f32->f16 convert (one copy per block).
// FEAT epilogue: fused exp-sum partials (shift=INV_TEMP) + target-score grab.
// !FEAT epilogue: store f32 scores.
// ---------------------------------------------------------------------------
template <bool FEAT>
__global__ void gemm_kernel(const _Float16* __restrict__ xh,
                            const float* __restrict__ Bsrc,
                            const int* __restrict__ targets,
                            float* __restrict__ sumexp,
                            float* __restrict__ pos,
                            float* __restrict__ kscore) {
  __shared__ _Float16 ldsB[BN * LDSP];
  const int tid   = threadIdx.x;
  const int lane  = tid & 31;
  const int wv    = tid >> 5;       // 0..7
  const int mbase = wv * 32;
  const int nbase = blockIdx.x * BN;
  const int lrow  = lane & 15;
  const int hi16  = lane >> 4;      // 0 | 1

  v8f acc[2][4];
#pragma unroll
  for (int mi = 0; mi < 2; ++mi)
#pragma unroll
    for (int nt = 0; nt < 4; ++nt)
#pragma unroll
      for (int j = 0; j < 8; ++j) acc[mi][nt][j] = 0.f;

  // LDS-fill addressing: each thread converts 8 floats of the 64x32 B slice
  const int fr = tid >> 2;          // tile row 0..63
  const int fk = (tid & 3) * 8;     // k offset 0,8,16,24

  for (int k0 = 0; k0 < D; k0 += 32) {
    const float4* g =
        (const float4*)(Bsrc + (size_t)(nbase + fr) * D + k0 + fk);
    float4 b0 = g[0], b1 = g[1];
    half8 hb;
    hb[0]=(_Float16)b0.x; hb[1]=(_Float16)b0.y;
    hb[2]=(_Float16)b0.z; hb[3]=(_Float16)b0.w;
    hb[4]=(_Float16)b1.x; hb[5]=(_Float16)b1.y;
    hb[6]=(_Float16)b1.z; hb[7]=(_Float16)b1.w;
    __syncthreads();                               // prior readers done
    *(half8*)&ldsB[fr * LDSP + fk] = hb;
    __syncthreads();                               // tile ready

    // A fragments (16x32 f16, ISA 7.12.2 layout): K chunks {klo, klo+16}
    const int klo = k0 + hi16 * 8;
    v16h afrag[2];
#pragma unroll
    for (int mi = 0; mi < 2; ++mi) {
      const _Float16* ap = xh + (size_t)(mbase + mi * 16 + lrow) * D + klo;
      afrag[mi] = combine16(*(const half8*)ap, *(const half8*)(ap + 16));
    }
#pragma unroll
    for (int nt = 0; nt < 4; ++nt) {
      // B fragment (32x16): col=lrow, K chunks {hi16*8, hi16*8+16}
      const _Float16* bp = &ldsB[(nt * 16 + lrow) * LDSP + hi16 * 8];
      v16h bfrag = combine16(*(const half8*)bp, *(const half8*)(bp + 16));
#pragma unroll
      for (int mi = 0; mi < 2; ++mi)
        acc[mi][nt] = __builtin_amdgcn_wmma_f32_16x16x32_f16(
            false, afrag[mi], false, bfrag, (short)0, acc[mi][nt],
            false, false);
    }
  }

  if (FEAT) {
    // C layout: VGPR j -> row mbase+mi*16+j+8*hi16, col nbase+nt*16+lrow
#pragma unroll
    for (int mi = 0; mi < 2; ++mi) {
#pragma unroll
      for (int j = 0; j < 8; ++j) {
        const int row = mbase + mi * 16 + j + hi16 * 8;
        const int tg  = targets[row];
        float v = 0.f;
#pragma unroll
        for (int nt = 0; nt < 4; ++nt) {
          const float o = acc[mi][nt][j];
          v += expf(fmaf(o, INV_TEMP, -INV_TEMP));   // exp(o/T - 20)
          if (tg == nbase + nt * 16 + lrow) pos[row] = o;
        }
#pragma unroll
        for (int m = 1; m <= 8; m <<= 1) v += __shfl_xor(v, m, 32);
        if (lrow == 0) atomicAdd(&sumexp[row], v);
      }
    }
  } else {
#pragma unroll
    for (int mi = 0; mi < 2; ++mi)
#pragma unroll
      for (int nt = 0; nt < 4; ++nt)
#pragma unroll
        for (int j = 0; j < 8; ++j) {
          const int row = mbase + mi * 16 + j + hi16 * 8;
          const int col = nbase + nt * 16 + lrow;
          kscore[(size_t)row * KCLUST + col] = acc[mi][nt][j];
        }
  }
}

// ---------------------------------------------------------------------------
// Kernel 3: per-row top-20 (own kmeans pid masked to -2) + 21-way CE at pos 0
// One block per batch row; the 32 KB score row is staged global->LDS by the
// Tensor Data Mover (tensor_load_to_lds, TENSORcnt), issued by wave 0 only.
// D# per CDNA5 ISA §8.3/8.4: 1-D tensor, data_size=4B, dim0=tile0=8192.
// ---------------------------------------------------------------------------
__global__ void topk_ce_kernel(const float* __restrict__ kscore,
                               const float* __restrict__ pos,
                               const int* __restrict__ kmeans_pids,
                               const int* __restrict__ indexes,
                               float* __restrict__ ce_neg) {
  const int b = blockIdx.x, t = threadIdx.x;
  __shared__ float s[KCLUST];
  __shared__ float wm[8];
  __shared__ int   wi[8];
  __shared__ float vals[20];

  if (t < 32) {  // one TDM issue per block (TDM ignores EXEC; per-wave issue)
    const uint64_t gaddr    = (uint64_t)(const void*)(kscore + (size_t)b * KCLUST);
    const uint32_t lds_base = (uint32_t)(uintptr_t)(void*)s;  // LDS byte offset
    u32x4 g0;
    g0[0] = 1u;                                   // count=1, user descriptor
    g0[1] = lds_base;                             // lds_addr
    g0[2] = (uint32_t)gaddr;                      // global_addr[31:0]
    g0[3] = 0x80000000u |                         // type=2 ("image")
            ((uint32_t)(gaddr >> 32) & 0x01FFFFFFu); // global_addr[56:32]
    u32x8 g1;
    g1[0] = 0x00020000u;          // workgroup_mask=0, data_size=2 (4 bytes)
    g1[1] = (uint32_t)KCLUST << 16;  // tensor_dim0[15:0] in [31:16]
    g1[2] = 1u << 16;             // tensor_dim0 hi=0 | tensor_dim1=1
    g1[3] = (uint32_t)KCLUST << 16;  // tensor_dim1 hi=0 | tile_dim0=8192
    g1[4] = 1u;                   // tile_dim1=1, tile_dim2=0 (unused)
    g1[5] = (uint32_t)KCLUST;     // tensor_dim0_stride[31:0]
    g1[6] = 0u;                   // stride hi | tensor_dim1_stride lo
    g1[7] = 0u;
    u32x4 gz = {0u, 0u, 0u, 0u};  // groups 2/3: dims/strides zero (unused)
    asm volatile("tensor_load_to_lds %0, %1, %2, %3\n\t"
                 "s_wait_tensorcnt 0x0"
                 :: "s"(g0), "s"(g1), "s"(gz), "s"(gz)
                 : "memory");
  }
  __syncthreads();                // LDS row visible to all waves

  if (t == 0) s[kmeans_pids[indexes[b]]] = -2.0f;  // scatter own cluster
  __syncthreads();
  const int lane = t & 31, wv = t >> 5;
  for (int it = 0; it < 20; ++it) {
    float m = -3.0f; int mi = 0;
    for (int i = t; i < KCLUST; i += 256) {
      const float v = s[i];
      if (v > m) { m = v; mi = i; }
    }
#pragma unroll
    for (int msk = 16; msk >= 1; msk >>= 1) {
      const float om = __shfl_xor(m, msk, 32);
      const int   oi = __shfl_xor(mi, msk, 32);
      if (om > m) { m = om; mi = oi; }
    }
    if (lane == 0) { wm[wv] = m; wi[wv] = mi; }
    __syncthreads();
    if (t == 0) {
      float bm = wm[0]; int bi = wi[0];
      for (int w = 1; w < 8; ++w) if (wm[w] > bm) { bm = wm[w]; bi = wi[w]; }
      vals[it] = bm;        // masked == original for selectable entries
      s[bi] = -2.0f;
    }
    __syncthreads();
  }
  if (t == 0) {
    float lg0 = pos[b] * INV_TEMP, mx = lg0;
    float lg[20];
    for (int i = 0; i < 20; ++i) {
      lg[i] = vals[i] * INV_TEMP;
      if (lg[i] > mx) mx = lg[i];
    }
    float sum = expf(lg0 - mx);
    for (int i = 0; i < 20; ++i) sum += expf(lg[i] - mx);
    ce_neg[b] = (logf(sum) + mx) - lg0;
  }
}

// ---------------------------------------------------------------------------
// Kernel 4: group modal-pid confidence mask + weighted means -> scalar loss
// ---------------------------------------------------------------------------
__global__ void finalize_kernel(const float* __restrict__ sumexp,
                                const float* __restrict__ pos,
                                const float* __restrict__ ce_neg,
                                const int* __restrict__ kmeans_pids,
                                const int* __restrict__ indexes,
                                float* __restrict__ out) {
  __shared__ int   pid[BATCH];
  __shared__ int   modev[8];
  __shared__ float red[BATCH];
  const int t = threadIdx.x;
  pid[t] = kmeans_pids[indexes[t]];
  __syncthreads();
  if (t < 8) {                         // mode over 16 pids; ties -> smallest
    int bc = 0, bp = 0x7fffffff;
    for (int i = 0; i < 16; ++i) {
      const int p = pid[t * 16 + i];
      int c = 0;
      for (int j = 0; j < 16; ++j) c += (pid[t * 16 + j] == p);
      if (c > bc || (c == bc && p < bp)) { bc = c; bp = p; }
    }
    modev[t] = bp;
  }
  __syncthreads();
  const int bh = (t < 128) ? t : (t - 128);
  const float mask = (pid[bh] == modev[bh >> 4]) ? 1.0f : 0.0f;
  // ce_main_b = (20 + log(sumexp_b)) - pos_b/T   (shift-20 logsumexp)
  const float ce_main = (INV_TEMP + logf(sumexp[t])) - pos[t] * INV_TEMP;
  red[t] = 0.2f * mask * ce_neg[t] + ce_main;
  __syncthreads();
  for (int s2 = 128; s2 >= 1; s2 >>= 1) {
    if (t < s2) red[t] += red[t + s2];
    __syncthreads();
  }
  if (t == 0) out[0] = red[0] / (float)BATCH;
}

// ---------------------------------------------------------------------------
extern "C" void kernel_launch(void* const* d_in, const int* in_sizes, int n_in,
                              void* d_out, int out_size, void* d_ws,
                              size_t ws_size, hipStream_t stream) {
  const float* inputs      = (const float*)d_in[0];
  const float* features    = (const float*)d_in[1];
  const float* centroids   = (const float*)d_in[2];
  const int*   targets     = (const int*)d_in[3];
  const int*   kmeans_pids = (const int*)d_in[4];
  const int*   indexes     = (const int*)d_in[5];
  float*       out         = (float*)d_out;
  (void)in_sizes; (void)n_in; (void)out_size; (void)ws_size;

  char* ws = (char*)d_ws;
  _Float16* xh     = (_Float16*)ws;                       // 1 MB
  float*    sumexp = (float*)(ws + (1 << 20));            // 1 KB
  float*    pos    = (float*)(ws + (1 << 20) + 1024);     // 1 KB
  float*    ce_neg = (float*)(ws + (1 << 20) + 2048);     // 1 KB
  float*    kscore = (float*)(ws + (2 << 20));            // 8 MB

  normalize_kernel<<<BATCH, 256, 0, stream>>>(inputs, xh, sumexp);
  gemm_kernel<true><<<NFEAT / BN, 256, 0, stream>>>(
      xh, features, targets, sumexp, pos, nullptr);
  gemm_kernel<false><<<KCLUST / BN, 256, 0, stream>>>(
      xh, centroids, nullptr, nullptr, nullptr, kscore);
  topk_ce_kernel<<<BATCH, 256, 0, stream>>>(kscore, pos, kmeans_pids,
                                            indexes, ce_neg);
  finalize_kernel<<<1, 256, 0, stream>>>(sumexp, pos, ce_neg, kmeans_pids,
                                         indexes, out);
}
